// GINConvNet_60095182406071
// MI455X (gfx1250) — matrix operand order repaired and checked by reading, not compiled, not run on hardware
//
#include <hip/hip_runtime.h>

typedef __attribute__((ext_vector_type(16))) _Float16 v16h;
typedef __attribute__((ext_vector_type(8)))  _Float16 v8h;
typedef __attribute__((ext_vector_type(8)))  float    v8f;

#define NN 200000
#define NG 1024
#define NE 3200000

__device__ __forceinline__ v8f wmma_f16(const v16h& a, const v16h& b, const v8f& c) {
  // D = A(16x32 f16) * B(32x16 f16) + C(16x16 f32)
  return __builtin_amdgcn_wmma_f32_16x16x32_f16(false, a, false, b, (short)0, c, false, false);
}

__device__ __forceinline__ v16h cat8(v8h lo, v8h hi) {
  return __builtin_shufflevector(lo, hi, 0,1,2,3,4,5,6,7,8,9,10,11,12,13,14,15);
}

// ---------------- utility ----------------
__global__ void k_zero(float* __restrict__ p, long n) {
  long i = (long)blockIdx.x * blockDim.x + threadIdx.x;
  long stride = (long)gridDim.x * blockDim.x;
  for (; i < n; i += stride) p[i] = 0.0f;
}

// B-operand swizzle: W[K x N] fp32 -> [K/32][N/16][lane 0..31][e 0..15] f16
// element: k = kcb*32 + e + ((e>=8)?8:0) + 8*(lane>>4), n = nt*16 + (lane&15); zero for k>=Kin
__global__ void k_swz_b(const float* __restrict__ w, _Float16* __restrict__ o,
                        int Kin, int K, int N) {
  int i = blockIdx.x * blockDim.x + threadIdx.x;
  int ntiles = N >> 4;
  int total = (K >> 5) * ntiles * 512;
  if (i >= total) return;
  int e    = i & 15;
  int lane = (i >> 4) & 31;
  int tile = i >> 9;
  int nt   = tile % ntiles;
  int kcb  = tile / ntiles;
  int half = lane >> 4, l16 = lane & 15;
  int k = (kcb << 5) + e + ((e >= 8) ? 8 : 0) + 8 * half;
  int n = (nt << 4) + l16;
  o[i] = (k < Kin) ? (_Float16)w[k * N + n] : (_Float16)0.0f;
}

// A-operand swizzle: W[M x K] fp32 -> [K/32][M/16][lane][e] f16
// element: k = kcb*32 + e + 16*(lane>>4), m = mt*16 + (lane&15)
__global__ void k_swz_a(const float* __restrict__ w, _Float16* __restrict__ o,
                        int M, int K) {
  int i = blockIdx.x * blockDim.x + threadIdx.x;
  int mtiles = M >> 4;
  int total = (K >> 5) * mtiles * 512;
  if (i >= total) return;
  int e    = i & 15;
  int lane = (i >> 4) & 31;
  int tile = i >> 9;
  int mt   = tile % mtiles;
  int kcb  = tile / mtiles;
  int half = lane >> 4, l16 = lane & 15;
  int k = (kcb << 5) + e + 16 * half;
  int m = (mt << 4) + l16;
  o[i] = (_Float16)w[(long)m * K + k];
}

// ---------------- GIN: edge scatter-add (agg[dst] += h[src]) ----------------
__global__ void k_scatter(const float* __restrict__ h, const int* __restrict__ ei,
                          float* __restrict__ agg, int C) {
  int e = blockIdx.x * blockDim.x + threadIdx.x;
  if (e >= NE) return;
  int src = ei[e];            // edge_index[0][e]
  int dst = ei[NE + e];       // edge_index[1][e]
  const float* hs = h + (long)src * C;
  float* ad = agg + (long)dst * C;
  if (C == 32) {
    const float4* h4 = (const float4*)hs;   // rows are 128B aligned
    #pragma unroll
    for (int j = 0; j < 8; ++j) {
      float4 v = h4[j];
      atomicAdd(ad + 4 * j + 0, v.x);
      atomicAdd(ad + 4 * j + 1, v.y);
      atomicAdd(ad + 4 * j + 2, v.z);
      atomicAdd(ad + 4 * j + 3, v.w);
    }
  } else {
    for (int c = 0; c < C; ++c) atomicAdd(ad + c, hs[c]);
  }
}

// ---------------- GIN: fused (h+agg) -> relu(W1)->relu(W2) + BN stats --------
// block = 256 threads = 8 waves; each wave owns a 16-node tile (128 nodes/block)
// w1s: swizzled B [Kpad/32][2][32][16]; w2s: swizzled B [1][2][32][16]
__global__ void k_gin_mlp(const float* __restrict__ hin, const float* __restrict__ agg,
                          const _Float16* __restrict__ w1s, const float* __restrict__ b1,
                          const _Float16* __restrict__ w2s, const float* __restrict__ b2,
                          float* __restrict__ z, float* __restrict__ stats,
                          int Kin, int Kpad) {
  __shared__ _Float16 zin[128 * 96];
  __shared__ _Float16 hid[128 * 32];

  int tid = threadIdx.x;
  long blockBase = (long)blockIdx.x * 128;

  int tot = 128 * Kpad;
  for (int i = tid; i < tot; i += 256) {
    int r = i / Kpad, k = i % Kpad;
    long node = blockBase + r;
    float v = 0.0f;
    if (k < Kin && node < NN) v = hin[node * Kin + k] + agg[node * Kin + k];
    zin[i] = (_Float16)v;
  }
  __syncthreads();

  int wave = tid >> 5;
  int lane = tid & 31;
  int half = lane >> 4;
  int l16  = lane & 15;

  const _Float16* zrow = zin + (wave * 16 + l16) * Kpad;

  v8f acc0 = {}; v8f acc1 = {};
  for (int kcb = 0; kcb < (Kpad >> 5); ++kcb) {
    v8h lo = *(const v8h*)(zrow + (kcb << 5) + 8 * half);        // 16B aligned
    v8h hi = *(const v8h*)(zrow + (kcb << 5) + 16 + 8 * half);
    v16h a = cat8(lo, hi);
    v16h bb0 = *(const v16h*)(w1s + (kcb * 2 + 0) * 512 + lane * 16);
    v16h bb1 = *(const v16h*)(w1s + (kcb * 2 + 1) * 512 + lane * 16);
    acc0 = wmma_f16(a, bb0, acc0);
    acc1 = wmma_f16(a, bb1, acc1);
  }

  _Float16* ht = hid + wave * 16 * 32;
  float bia0 = b1[l16], bia1 = b1[16 + l16];
  #pragma unroll
  for (int r = 0; r < 8; ++r) {
    int m = r + 8 * half;                                // C/D layout
    float v0 = acc0[r] + bia0; v0 = v0 > 0.0f ? v0 : 0.0f;
    float v1 = acc1[r] + bia1; v1 = v1 > 0.0f ? v1 : 0.0f;
    ht[m * 32 + l16]      = (_Float16)v0;
    ht[m * 32 + 16 + l16] = (_Float16)v1;
  }
  __syncthreads();

  v8f c0 = {}; v8f c1 = {};
  {
    const _Float16* hrow = ht + l16 * 32;
    v8h lo = *(const v8h*)(hrow + 8 * half);
    v8h hi = *(const v8h*)(hrow + 16 + 8 * half);
    v16h a = cat8(lo, hi);
    v16h bb0 = *(const v16h*)(w2s + 0 * 512 + lane * 16);
    v16h bb1 = *(const v16h*)(w2s + 1 * 512 + lane * 16);
    c0 = wmma_f16(a, bb0, c0);
    c1 = wmma_f16(a, bb1, c1);
  }

  float b20 = b2[l16], b21 = b2[16 + l16];
  float s0 = 0, ss0 = 0, s1 = 0, ss1 = 0;
  #pragma unroll
  for (int r = 0; r < 8; ++r) {
    int m = r + 8 * half;
    long node = blockBase + wave * 16 + m;
    float v0 = c0[r] + b20; v0 = v0 > 0.0f ? v0 : 0.0f;
    float v1 = c1[r] + b21; v1 = v1 > 0.0f ? v1 : 0.0f;
    if (node < NN) {
      z[node * 32 + l16]      = v0;
      z[node * 32 + 16 + l16] = v1;
      s0 += v0; ss0 += v0 * v0;
      s1 += v1; ss1 += v1 * v1;
    }
  }
  atomicAdd(stats + l16, s0);
  atomicAdd(stats + 32 + l16, ss0);
  atomicAdd(stats + 16 + l16, s1);
  atomicAdd(stats + 48 + l16, ss1);
}

// ---------------- GIN: batchnorm apply (z -> h) + zero agg for next layer ----
__global__ void k_bn_apply(const float* __restrict__ z, const float* __restrict__ stats,
                           const float* __restrict__ gamma, const float* __restrict__ beta,
                           float* __restrict__ h, float* __restrict__ agg) {
  long i = (long)blockIdx.x * blockDim.x + threadIdx.x;
  long n = (long)NN * 32;
  long stride = (long)gridDim.x * blockDim.x;
  const float inv = 1.0f / (float)NN;
  for (; i < n; i += stride) {
    int c = (int)(i & 31);
    float mu  = stats[c] * inv;
    float var = stats[32 + c] * inv - mu * mu;
    float g = gamma[c] * rsqrtf(var + 1e-5f);
    h[i] = (z[i] - mu) * g + beta[c];
    agg[i] = 0.0f;
  }
}

// ---------------- global add pool ----------------
__global__ void k_pool(const float* __restrict__ h, const int* __restrict__ batch,
                       float* __restrict__ pooled) {
  int node = blockIdx.x * blockDim.x + threadIdx.x;
  if (node >= NN) return;
  int g = batch[node];
  const float4* h4 = (const float4*)(h + (long)node * 32);
  float* pd = pooled + g * 32;
  #pragma unroll
  for (int j = 0; j < 8; ++j) {
    float4 v = h4[j];
    atomicAdd(pd + 4 * j + 0, v.x);
    atomicAdd(pd + 4 * j + 1, v.y);
    atomicAdd(pd + 4 * j + 2, v.z);
    atomicAdd(pd + 4 * j + 3, v.w);
  }
}

// ---------------- fused embed-gather + Conv1d as WMMA GEMM -------------------
// one block per graph b; out[b,o,t] = sum_{c,k} embed[target[b,c]][t+k]*W[o,c*8+k]
// Wswz: A-swizzled [250][2][32][16]; M=32 (2 m-tiles), N=128 (8 waves x 16), K=8000
__global__ void k_conv(const _Float16* __restrict__ Wswz, const int* __restrict__ target,
                       const float* __restrict__ embed, const float* __restrict__ cb,
                       float* __restrict__ out) {
  __shared__ int tg[1000];
  __shared__ _Float16 em[26 * 128];
  int b = blockIdx.x;
  int tid = threadIdx.x;
  for (int i = tid; i < 1000; i += 256) tg[i] = target[b * 1000 + i];
  for (int i = tid; i < 26 * 128; i += 256) em[i] = (_Float16)embed[i];
  __syncthreads();

  int wave = tid >> 5, lane = tid & 31, half = lane >> 4, l16 = lane & 15;
  int t = wave * 16 + l16;
  bool valid = (t <= 120);

  v8f acc0 = {}; v8f acc1 = {};
  for (int kcb = 0; kcb < 250; ++kcb) {
    v16h a0 = *(const v16h*)(Wswz + (kcb * 2 + 0) * 512 + lane * 16);
    v16h a1 = *(const v16h*)(Wswz + (kcb * 2 + 1) * 512 + lane * 16);
    v16h bb;
    int c0 = kcb * 4 + 2 * half;          // K = kcb*32 + 16*half + e ; c = K>>3
    if (valid) {
      const _Float16* r0 = em + tg[c0] * 128 + t;
      const _Float16* r1 = em + tg[c0 + 1] * 128 + t;
      #pragma unroll
      for (int e = 0; e < 8; ++e) { bb[e] = r0[e]; bb[e + 8] = r1[e]; }
    } else {
      #pragma unroll
      for (int e = 0; e < 16; ++e) bb[e] = (_Float16)0.0f;
    }
    acc0 = wmma_f16(a0, bb, acc0);
    acc1 = wmma_f16(a1, bb, acc1);
  }
  if (valid) {
    #pragma unroll
    for (int r = 0; r < 8; ++r) {
      int o0 = r + 8 * half;
      int o1 = 16 + o0;
      out[(long)b * 3872 + o0 * 121 + t] = acc0[r] + cb[o0];
      out[(long)b * 3872 + o1 * 121 + t] = acc1[r] + cb[o1];
    }
  }
}

// ---------------- generic WMMA GEMM: C = act(A[MxK] @ W[KxN] + bias) ---------
// block = 128 threads (4 waves): 16 rows x 64 cols; no LDS, no barriers.
// Wswz: B-swizzled [K/32][N/16][32][16]. M%16==0, N%64==0, K%32==0.
__global__ void k_gemm(const float* __restrict__ A, const _Float16* __restrict__ Wswz,
                       const float* __restrict__ bias, float* __restrict__ C,
                       int N, int K, int lda, int ldc, int relu) {
  int tid = threadIdx.x;
  int wave = tid >> 5, lane = tid & 31, half = lane >> 4, l16 = lane & 15;
  int mBase = blockIdx.x * 16;
  int ntile = blockIdx.y * 4 + wave;
  int ntiles = N >> 4;
  const float* arow = A + (long)(mBase + l16) * lda;

  v8f acc = {};
  for (int kcb = 0; kcb < (K >> 5); ++kcb) {
    const float* p = arow + (kcb << 5) + 8 * half;   // 32B aligned segments
    v16h a;
    #pragma unroll
    for (int e = 0; e < 8; ++e) {
      a[e]     = (_Float16)p[e];
      a[e + 8] = (_Float16)p[16 + e];
    }
    v16h bb = *(const v16h*)(Wswz + ((long)kcb * ntiles + ntile) * 512 + lane * 16);
    acc = wmma_f16(a, bb, acc);
  }
  int n = ntile * 16 + l16;
  float bv = bias[n];
  #pragma unroll
  for (int r = 0; r < 8; ++r) {
    int m = mBase + r + 8 * half;
    float v = acc[r] + bv;
    if (relu) v = v > 0.0f ? v : 0.0f;
    C[(long)m * ldc + n] = v;
  }
}

// ---------------- final [1024,256]@[256,1] ----------------
__global__ void k_out(const float* __restrict__ A, const float* __restrict__ w,
                      const float* __restrict__ b, float* __restrict__ out) {
  int row = blockIdx.x * blockDim.x + threadIdx.x;
  if (row >= NG) return;
  float s = b[0];
  for (int k = 0; k < 256; ++k) s += A[row * 256 + k] * w[k];
  out[row] = s;
}

extern "C" void kernel_launch(void* const* d_in, const int* in_sizes, int n_in,
                              void* d_out, int out_size, void* d_ws, size_t ws_size,
                              hipStream_t stream) {
  (void)in_sizes; (void)n_in; (void)out_size; (void)ws_size;
  const float* x      = (const float*)d_in[0];
  const int*   ei     = (const int*)d_in[1];
  const int*   batch  = (const int*)d_in[2];
  const int*   target = (const int*)d_in[3];
  const float* g1w1   = (const float*)d_in[4];
  const float* g1b1   = (const float*)d_in[5];
  const float* g1w2   = (const float*)d_in[6];
  const float* g1b2   = (const float*)d_in[7];
  const float* gW1    = (const float*)d_in[8];
  const float* gb1    = (const float*)d_in[9];
  const float* gW2    = (const float*)d_in[10];
  const float* gb2    = (const float*)d_in[11];
  const float* bng    = (const float*)d_in[12];
  const float* bnb    = (const float*)d_in[13];
  const float* fcxd_w = (const float*)d_in[14];
  const float* fcxd_b = (const float*)d_in[15];
  const float* embed  = (const float*)d_in[16];
  const float* conv_w = (const float*)d_in[17];
  const float* conv_b = (const float*)d_in[18];
  const float* fcxt_w = (const float*)d_in[19];
  const float* fcxt_b = (const float*)d_in[20];
  const float* fc1_w  = (const float*)d_in[21];
  const float* fc1_b  = (const float*)d_in[22];
  const float* fc2_w  = (const float*)d_in[23];
  const float* fc2_b  = (const float*)d_in[24];
  const float* out_w  = (const float*)d_in[25];
  const float* out_b  = (const float*)d_in[26];

  char* wsp = (char*)d_ws;
  auto alloc = [&](size_t bytes) {
    char* p = wsp; wsp += (bytes + 255) & ~(size_t)255; return p;
  };
  float*    agg    = (float*)alloc((size_t)NN * 78 * 4);
  float*    zbuf   = (float*)alloc((size_t)NN * 32 * 4);
  float*    hbuf   = (float*)alloc((size_t)NN * 32 * 4);
  float*    stats  = (float*)alloc(64 * 4);
  _Float16* w1h    = (_Float16*)alloc((size_t)3 * 2 * 512 * 2);      // up to Kpad=96
  _Float16* w2h    = (_Float16*)alloc((size_t)1 * 2 * 512 * 2);
  _Float16* cWh    = (_Float16*)alloc((size_t)250 * 2 * 512 * 2);    // conv A swz
  _Float16* fcxdS  = (_Float16*)alloc((size_t)1 * 8 * 512 * 2);
  _Float16* fcxtS  = (_Float16*)alloc((size_t)121 * 8 * 512 * 2);
  _Float16* fc1S   = (_Float16*)alloc((size_t)8 * 64 * 512 * 2);
  _Float16* fc2S   = (_Float16*)alloc((size_t)32 * 16 * 512 * 2);
  float*    convo  = (float*)alloc((size_t)NG * 3872 * 4);
  float*    pooled = (float*)alloc((size_t)NG * 32 * 4);
  float*    xc     = (float*)alloc((size_t)NG * 256 * 4);
  float*    f1o    = (float*)alloc((size_t)NG * 1024 * 4);
  float*    f2o    = (float*)alloc((size_t)NG * 256 * 4);

  // zero aggregation buffer (widest layout: 78 channels) once
  k_zero<<<2048, 256, 0, stream>>>(agg, (long)NN * 78);

  int mlpBlocks = (NN + 127) / 128;
  for (int layer = 0; layer < 5; ++layer) {
    const float* W1 = (layer == 0) ? g1w1 : gW1 + (layer - 1) * 1024;
    const float* B1 = (layer == 0) ? g1b1 : gb1 + (layer - 1) * 32;
    const float* W2 = (layer == 0) ? g1w2 : gW2 + (layer - 1) * 1024;
    const float* B2 = (layer == 0) ? g1b2 : gb2 + (layer - 1) * 32;
    int Kin  = (layer == 0) ? 78 : 32;
    int Kpad = (layer == 0) ? 96 : 32;
    const float* hin = (layer == 0) ? x : hbuf;

    k_zero<<<1, 64, 0, stream>>>(stats, 64);
    k_swz_b<<<((Kpad >> 5) * 2 * 512 + 255) / 256, 256, 0, stream>>>(W1, w1h, Kin, Kpad, 32);
    k_swz_b<<<(1 * 2 * 512 + 255) / 256, 256, 0, stream>>>(W2, w2h, 32, 32, 32);
    k_scatter<<<(NE + 255) / 256, 256, 0, stream>>>(hin, ei, agg, Kin);
    k_gin_mlp<<<mlpBlocks, 256, 0, stream>>>(hin, agg, w1h, B1, w2h, B2,
                                             zbuf, stats, Kin, Kpad);
    k_bn_apply<<<2048, 256, 0, stream>>>(zbuf, stats, bng + layer * 32,
                                         bnb + layer * 32, hbuf, agg);
  }

  // pooled -> xd (into xc columns [0,128))
  k_zero<<<(NG * 32 + 255) / 256, 256, 0, stream>>>(pooled, (long)NG * 32);
  k_pool<<<(NN + 255) / 256, 256, 0, stream>>>(hbuf, batch, pooled);
  k_swz_b<<<(8 * 512 + 255) / 256, 256, 0, stream>>>(fcxd_w, fcxdS, 32, 32, 128);
  {
    dim3 g(NG / 16, 128 / 64);
    k_gemm<<<g, 128, 0, stream>>>(pooled, fcxdS, fcxd_b, xc, 128, 32, 32, 256, 1);
  }

  // protein branch: conv as fused-gather WMMA GEMM, then fcxt into xc[:,128:256)
  k_swz_a<<<(250 * 2 * 512 + 255) / 256, 256, 0, stream>>>(conv_w, cWh, 32, 8000);
  k_conv<<<NG, 256, 0, stream>>>(cWh, target, embed, conv_b, convo);
  k_swz_b<<<(121 * 8 * 512 + 255) / 256, 256, 0, stream>>>(fcxt_w, fcxtS, 3872, 3872, 128);
  {
    dim3 g(NG / 16, 128 / 64);
    k_gemm<<<g, 128, 0, stream>>>(convo, fcxtS, fcxt_b, xc + 128, 128, 3872, 3872, 256, 0);
  }

  // dense head
  k_swz_b<<<(8 * 64 * 512 + 255) / 256, 256, 0, stream>>>(fc1_w, fc1S, 256, 256, 1024);
  {
    dim3 g(NG / 16, 1024 / 64);
    k_gemm<<<g, 128, 0, stream>>>(xc, fc1S, fc1_b, f1o, 1024, 256, 256, 1024, 1);
  }
  k_swz_b<<<(32 * 16 * 512 + 255) / 256, 256, 0, stream>>>(fc2_w, fc2S, 1024, 1024, 256);
  {
    dim3 g(NG / 16, 256 / 64);
    k_gemm<<<g, 128, 0, stream>>>(f1o, fc2S, fc2_b, f2o, 256, 1024, 1024, 256, 0);
  }
  k_out<<<(NG + 255) / 256, 256, 0, stream>>>(f2o, out_w, out_b, (float*)d_out);
}
